// CausalAttention_6983616824221
// MI455X (gfx1250) — matrix operand order repaired and checked by reading, not compiled, not run on hardware
//
#include <hip/hip_runtime.h>

typedef __bf16 bf16;
typedef __attribute__((ext_vector_type(16))) __bf16 v16bf;
typedef __attribute__((ext_vector_type(8)))  float  v8f;

#define NB      8
#define S_LEN   2048
#define DIM     1024
#define LDK     32
#define LDS_ST  40               // padded LDS row stride (elements, 80B)
#define INV_SQRT_D (1.0f/32.0f)
#define NEG_INF (-__builtin_inff())

// ---------- CDNA5 async global->LDS copy (ASYNCcnt path) -----------------

__device__ __forceinline__ void async_b128(bf16* lds_dst, const bf16* src) {
  // LDS generic pointer: low 32 bits are the LDS byte offset (ISA 10.2).
  unsigned dst = (unsigned)(size_t)lds_dst;
  unsigned long long ga = (unsigned long long)(const void*)src;
  asm volatile("global_load_async_to_lds_b128 %0, %1, off"
               :: "v"(dst), "v"(ga) : "memory");
}

__device__ __forceinline__ void wait_async() {
  asm volatile("s_wait_asynccnt 0x0" ::: "memory");
}

// ---------- WMMA helpers -------------------------------------------------

__device__ __forceinline__ v8f wmma_bf16(v16bf a, v16bf b, v8f c) {
  return __builtin_amdgcn_wmma_f32_16x16x32_bf16(
      false, a, false, b, (short)0, c, false, false);
}

// A fragment: lane half h=lane>>4: K = {h*8+e (e<8), 16+h*8+(e-8)}
__device__ __forceinline__ v16bf frag_a(const bf16* tile, int row, int lane) {
  const bf16* p = tile + row * LDS_ST + ((lane >> 4) << 3);
  v16bf f;
#pragma unroll
  for (int e = 0; e < 8; ++e) f[e] = p[e];
#pragma unroll
  for (int e = 0; e < 8; ++e) f[e + 8] = p[e + 16];
  return f;
}

// B fragment (tile stored as B^T: [n][k]): K = {h*8+e (e<8), 16+h*8+(e-8)}
__device__ __forceinline__ v16bf frag_b(const bf16* tile, int row, int lane) {
  const bf16* p = tile + row * LDS_ST + ((lane >> 4) << 3);
  v16bf f;
#pragma unroll
  for (int e = 0; e < 8; ++e) f[e] = p[e];
#pragma unroll
  for (int e = 0; e < 8; ++e) f[e + 8] = p[e + 16];
  return f;
}

// ---------- shared GEMM core: C(128x128) += A(128xK) * B^T(128xK) --------
// Both operands row-major with contiguous K; tiles filled via async DMA.

__device__ __forceinline__ void gemm_tiles(
    const bf16* __restrict__ aBase, size_t aStride,
    const bf16* __restrict__ bBase, size_t bStride,
    int ksize, int tid, bf16* ldsA, bf16* ldsB, v8f (&acc)[2][4]) {
  const int lane = tid & 31, wave = tid >> 5;
  const int rb = wave & 3, cb = wave >> 2;
  const int c0 = tid, c1 = tid + 256;           // 512 16B chunks per tile
  const int m0c = c0 >> 2, k0c = (c0 & 3) * 8;
  const int m1c = c1 >> 2, k1c = (c1 & 3) * 8;

  for (int kk = 0; kk < ksize; kk += LDK) {
    async_b128(ldsA + m0c * LDS_ST + k0c, aBase + (size_t)m0c * aStride + kk + k0c);
    async_b128(ldsA + m1c * LDS_ST + k1c, aBase + (size_t)m1c * aStride + kk + k1c);
    async_b128(ldsB + m0c * LDS_ST + k0c, bBase + (size_t)m0c * bStride + kk + k0c);
    async_b128(ldsB + m1c * LDS_ST + k1c, bBase + (size_t)m1c * bStride + kk + k1c);
    if (kk + LDK < ksize) {
      __builtin_prefetch(aBase + (size_t)(tid >> 1) * aStride + kk + LDK, 0, 1);
      __builtin_prefetch(bBase + (size_t)(tid >> 1) * bStride + kk + LDK, 0, 1);
    }
    wait_async();
    __syncthreads();
    v16bf bfr[4];
#pragma unroll
    for (int nt = 0; nt < 4; ++nt)
      bfr[nt] = frag_b(ldsB, cb * 64 + nt * 16 + (lane & 15), lane);
#pragma unroll
    for (int mt = 0; mt < 2; ++mt) {
      v16bf afr = frag_a(ldsA, rb * 32 + mt * 16 + (lane & 15), lane);
#pragma unroll
      for (int nt = 0; nt < 4; ++nt)
        acc[mt][nt] = wmma_bf16(afr, bfr[nt], acc[mt][nt]);
    }
    __syncthreads();
  }
}

// ---------- Kernel 0: init partial-reduction buffers ---------------------

__global__ void init_parts(float* __restrict__ cmax_part,
                           float* __restrict__ csum_part, int n) {
  int t = blockIdx.x * 256 + threadIdx.x;
  if (t < n) { cmax_part[t] = NEG_INF; csum_part[t] = 0.0f; }
}

// ---------- Kernel 1: x (f32) -> xb (bf16), 4 elements/thread ------------

__global__ void convert_x(const float* __restrict__ x, bf16* __restrict__ xb) {
  size_t t = ((size_t)blockIdx.x * 256 + threadIdx.x) * 4;
  if (t >= (size_t)NB * S_LEN * DIM) return;
  float4 f = *(const float4*)(x + t);
  xb[t + 0] = (bf16)f.x; xb[t + 1] = (bf16)f.y;
  xb[t + 2] = (bf16)f.z; xb[t + 3] = (bf16)f.w;
}

// ---------- Kernel 2: W (f32, [k][n]) -> W^T (bf16, [n][k]) --------------

__global__ void transpose_w(
    const float* __restrict__ Wq, const float* __restrict__ Wk,
    const float* __restrict__ Wv,
    bf16* __restrict__ WqT, bf16* __restrict__ WkT, bf16* __restrict__ WvT) {
  __shared__ float tile[32][33];
  const int z = blockIdx.z;
  const float* W = (z == 0) ? Wq : (z == 1) ? Wk : Wv;
  bf16*       T = (z == 0) ? WqT : (z == 1) ? WkT : WvT;
  const int k0 = blockIdx.x * 32, n0 = blockIdx.y * 32;
  const int tx = threadIdx.x, ty = threadIdx.y;
#pragma unroll
  for (int i = 0; i < 4; ++i)
    tile[ty + 8 * i][tx] = W[(size_t)(k0 + ty + 8 * i) * DIM + n0 + tx];
  __syncthreads();
#pragma unroll
  for (int i = 0; i < 4; ++i)
    T[(size_t)(n0 + ty + 8 * i) * DIM + k0 + tx] = (bf16)tile[tx][ty + 8 * i];
}

// ---------- Kernel 3: fused QKV projection -------------------------------

__global__ __launch_bounds__(256) void qkv_gemm(
    const bf16* __restrict__ xb,
    const bf16* __restrict__ WqT, const bf16* __restrict__ WkT,
    const bf16* __restrict__ WvT,
    bf16* __restrict__ Qb, bf16* __restrict__ Kb, bf16* __restrict__ Vb) {
  __shared__ bf16 ldsA[128 * LDS_ST];
  __shared__ bf16 ldsB[128 * LDS_ST];
  const int tid = threadIdx.x;
  const int lane = tid & 31, wave = tid >> 5;
  const int rb = wave & 3, cb = wave >> 2;
  const int m0 = blockIdx.x * 128;
  const int n0 = blockIdx.y * 128;
  const int wsel = blockIdx.z;
  const bf16* WT = (wsel == 0) ? WqT : (wsel == 1) ? WkT : WvT;
  bf16*      Out = (wsel == 0) ? Qb  : (wsel == 1) ? Kb  : Vb;

  v8f acc[2][4] = {};
  gemm_tiles(xb + (size_t)m0 * DIM, DIM, WT + (size_t)n0 * DIM, DIM,
             DIM, tid, ldsA, ldsB, acc);

#pragma unroll
  for (int mt = 0; mt < 2; ++mt)
#pragma unroll
    for (int nt = 0; nt < 4; ++nt) {
      int row0 = m0 + rb * 32 + mt * 16 + ((lane >= 16) ? 8 : 0);
      int col  = n0 + cb * 64 + nt * 16 + (lane & 15);
#pragma unroll
      for (int r = 0; r < 8; ++r)
        Out[(size_t)(row0 + r) * DIM + col] = (bf16)acc[mt][nt][r];
    }
}

// ---------- Kernel 4: per-column (key-axis j) max partials ---------------

__global__ __launch_bounds__(256) void score_max(
    const bf16* __restrict__ Qb, const bf16* __restrict__ Kb,
    float* __restrict__ cmax_part) {
  const int it_i = blockIdx.x, it_j = blockIdx.y, b = blockIdx.z;
  if (it_j > it_i) return;                     // fully masked tile
  __shared__ bf16 ldsA[128 * LDS_ST];
  __shared__ bf16 ldsB[128 * LDS_ST];
  __shared__ float red[4][128];
  const int tid = threadIdx.x;
  const int lane = tid & 31, wave = tid >> 5;
  const int rb = wave & 3, cb = wave >> 2;
  const int i0 = it_i * 128, j0 = it_j * 128;

  v8f acc[2][4] = {};
  gemm_tiles(Qb + ((size_t)b * S_LEN + i0) * DIM, DIM,
             Kb + ((size_t)b * S_LEN + j0) * DIM, DIM,
             DIM, tid, ldsA, ldsB, acc);

#pragma unroll
  for (int nt = 0; nt < 4; ++nt) {
    int j = j0 + cb * 64 + nt * 16 + (lane & 15);
    float m = NEG_INF;
#pragma unroll
    for (int mt = 0; mt < 2; ++mt) {
      int ib = i0 + rb * 32 + mt * 16 + ((lane >= 16) ? 8 : 0);
#pragma unroll
      for (int r = 0; r < 8; ++r)
        if (j <= ib + r) m = fmaxf(m, acc[mt][nt][r] * INV_SQRT_D);
    }
    m = fmaxf(m, __shfl_xor(m, 16, 32));
    if (lane < 16) red[rb][cb * 64 + nt * 16 + lane] = m;
  }
  __syncthreads();
  if (tid < 128) {
    float m = fmaxf(fmaxf(red[0][tid], red[1][tid]),
                    fmaxf(red[2][tid], red[3][tid]));
    cmax_part[((size_t)b * 16 + it_i) * S_LEN + j0 + tid] = m;
  }
}

// ---------- Kernel 5: colmax over i-tiles --------------------------------

__global__ void reduce_max(const float* __restrict__ part,
                           float* __restrict__ colmax) {
  int t = blockIdx.x * 256 + threadIdx.x;
  if (t >= NB * S_LEN) return;
  int b = t / S_LEN, j = t % S_LEN;
  float m = NEG_INF;
#pragma unroll
  for (int it = 0; it < 16; ++it)
    m = fmaxf(m, part[((size_t)b * 16 + it) * S_LEN + j]);
  colmax[t] = m;
}

// ---------- Kernel 6: P = exp(s'-m[j]) (bf16) + column-sum partials ------

__global__ __launch_bounds__(256) void score_softmax(
    const bf16* __restrict__ Qb, const bf16* __restrict__ Kb,
    const float* __restrict__ colmax, bf16* __restrict__ Pb,
    float* __restrict__ csum_part) {
  const int it_i = blockIdx.x, it_j = blockIdx.y, b = blockIdx.z;
  if (it_j > it_i) return;
  __shared__ bf16 ldsA[128 * LDS_ST];
  __shared__ bf16 ldsB[128 * LDS_ST];
  __shared__ float red[4][128];
  const int tid = threadIdx.x;
  const int lane = tid & 31, wave = tid >> 5;
  const int rb = wave & 3, cb = wave >> 2;
  const int i0 = it_i * 128, j0 = it_j * 128;

  v8f acc[2][4] = {};
  gemm_tiles(Qb + ((size_t)b * S_LEN + i0) * DIM, DIM,
             Kb + ((size_t)b * S_LEN + j0) * DIM, DIM,
             DIM, tid, ldsA, ldsB, acc);

#pragma unroll
  for (int nt = 0; nt < 4; ++nt) {
    int j = j0 + cb * 64 + nt * 16 + (lane & 15);
    float mj = colmax[(size_t)b * S_LEN + j];
    float ssum = 0.0f;
#pragma unroll
    for (int mt = 0; mt < 2; ++mt) {
      int ib = i0 + rb * 32 + mt * 16 + ((lane >= 16) ? 8 : 0);
#pragma unroll
      for (int r = 0; r < 8; ++r) {
        int i = ib + r;
        float p = (j <= i) ? __expf(acc[mt][nt][r] * INV_SQRT_D - mj) : 0.0f;
        Pb[((size_t)b * S_LEN + i) * S_LEN + j] = (bf16)p;
        ssum += p;
      }
    }
    ssum += __shfl_xor(ssum, 16, 32);
    if (lane < 16) red[rb][cb * 64 + nt * 16 + lane] = ssum;
  }
  __syncthreads();
  if (tid < 128) {
    float s = red[0][tid] + red[1][tid] + red[2][tid] + red[3][tid];
    csum_part[((size_t)b * 16 + it_i) * S_LEN + j0 + tid] = s;
  }
}

// ---------- Kernel 7: 1/colsum ------------------------------------------

__global__ void reduce_sum(const float* __restrict__ part,
                           float* __restrict__ colrsum) {
  int t = blockIdx.x * 256 + threadIdx.x;
  if (t >= NB * S_LEN) return;
  int b = t / S_LEN, j = t % S_LEN;
  float s = 0.0f;
#pragma unroll
  for (int it = 0; it < 16; ++it)
    s += part[((size_t)b * 16 + it) * S_LEN + j];
  colrsum[t] = 1.0f / s;
}

// ---------- Kernel 8: VsT[b][e][j] = V[b][j][e] / colsum[b][j] -----------

__global__ void scale_v_t(const bf16* __restrict__ Vb,
                          const float* __restrict__ colrsum,
                          bf16* __restrict__ VsT) {
  __shared__ float tile[32][33];
  const int b = blockIdx.z;
  const int j0 = blockIdx.x * 32, e0 = blockIdx.y * 32;
  const int tx = threadIdx.x, ty = threadIdx.y;
#pragma unroll
  for (int i = 0; i < 4; ++i) {
    int j = j0 + ty + 8 * i;
    tile[ty + 8 * i][tx] =
        (float)Vb[((size_t)b * S_LEN + j) * DIM + e0 + tx] *
        colrsum[(size_t)b * S_LEN + j];
  }
  __syncthreads();
#pragma unroll
  for (int i = 0; i < 4; ++i)
    VsT[((size_t)b * DIM + e0 + ty + 8 * i) * S_LEN + j0 + tx] =
        (bf16)tile[tx][ty + 8 * i];
}

// ---------- Kernel 9: out = P @ Vs (bf16 WMMA, fp32 out) -----------------

__global__ __launch_bounds__(256) void out_gemm(
    const bf16* __restrict__ Pb, const bf16* __restrict__ VsT,
    float* __restrict__ out) {
  const int it_i = blockIdx.x, e_t = blockIdx.y, b = blockIdx.z;
  __shared__ bf16 ldsA[128 * LDS_ST];
  __shared__ bf16 ldsB[128 * LDS_ST];
  const int tid = threadIdx.x;
  const int lane = tid & 31, wave = tid >> 5;
  const int rb = wave & 3, cb = wave >> 2;
  const int i0 = it_i * 128, e0 = e_t * 128;
  const int jmax = i0 + 128;                   // P[i][j]==0 for j>i

  v8f acc[2][4] = {};
  gemm_tiles(Pb + ((size_t)b * S_LEN + i0) * S_LEN, S_LEN,
             VsT + ((size_t)b * DIM + e0) * S_LEN, S_LEN,
             jmax, tid, ldsA, ldsB, acc);

#pragma unroll
  for (int mt = 0; mt < 2; ++mt)
#pragma unroll
    for (int nt = 0; nt < 4; ++nt) {
      int row0 = i0 + rb * 32 + mt * 16 + ((lane >= 16) ? 8 : 0);
      int col  = e0 + cb * 64 + nt * 16 + (lane & 15);
#pragma unroll
      for (int r = 0; r < 8; ++r)
        out[((size_t)b * S_LEN + row0 + r) * DIM + col] = acc[mt][nt][r];
    }
}

// ---------- launcher -----------------------------------------------------

extern "C" void kernel_launch(void* const* d_in, const int* in_sizes, int n_in,
                              void* d_out, int out_size, void* d_ws,
                              size_t ws_size, hipStream_t stream) {
  const float* x  = (const float*)d_in[0];
  const float* Wq = (const float*)d_in[1];
  const float* Wk = (const float*)d_in[2];
  const float* Wv = (const float*)d_in[3];
  float* out = (float*)d_out;

  constexpr size_t SZ_XB   = (size_t)NB * S_LEN * DIM * sizeof(bf16);   // 32MB
  constexpr size_t SZ_W    = (size_t)DIM * DIM * sizeof(bf16);         // 2MB
  constexpr size_t SZ_P    = (size_t)NB * S_LEN * S_LEN * sizeof(bf16);// 64MB
  constexpr size_t SZ_PART = (size_t)NB * 16 * S_LEN * sizeof(float);  // 1MB
  constexpr size_t SZ_COL  = (size_t)NB * S_LEN * sizeof(float);       // 64KB
  char* ws = (char*)d_ws;
  size_t o = 0;
  bf16*  xb       = (bf16*)(ws + o); o += SZ_XB;
  bf16*  WqT      = (bf16*)(ws + o); o += SZ_W;
  bf16*  WkT      = (bf16*)(ws + o); o += SZ_W;
  bf16*  WvT      = (bf16*)(ws + o); o += SZ_W;
  bf16*  Qb       = (bf16*)(ws + o); o += SZ_XB;
  bf16*  Kb       = (bf16*)(ws + o); o += SZ_XB;
  bf16*  Vb       = (bf16*)(ws + o); o += SZ_XB;
  bf16*  VsT      = (bf16*)(ws + o); o += SZ_XB;
  bf16*  Pb       = (bf16*)(ws + o); o += SZ_P;
  float* cmaxPart = (float*)(ws + o); o += SZ_PART;
  float* csumPart = (float*)(ws + o); o += SZ_PART;
  float* colmax   = (float*)(ws + o); o += SZ_COL;
  float* colrsum  = (float*)(ws + o); o += SZ_COL;

  const int nPart = NB * 16 * S_LEN;
  init_parts<<<(nPart + 255) / 256, 256, 0, stream>>>(cmaxPart, csumPart,
                                                      nPart);
  convert_x<<<(unsigned)(((size_t)NB * S_LEN * DIM / 4 + 255) / 256), 256, 0,
              stream>>>(x, xb);
  transpose_w<<<dim3(32, 32, 3), dim3(32, 8), 0, stream>>>(Wq, Wk, Wv, WqT,
                                                           WkT, WvT);

  qkv_gemm<<<dim3(128, 8, 3), 256, 0, stream>>>(xb, WqT, WkT, WvT, Qb, Kb, Vb);

  score_max<<<dim3(16, 16, NB), 256, 0, stream>>>(Qb, Kb, cmaxPart);
  reduce_max<<<(NB * S_LEN + 255) / 256, 256, 0, stream>>>(cmaxPart, colmax);

  score_softmax<<<dim3(16, 16, NB), 256, 0, stream>>>(Qb, Kb, colmax, Pb,
                                                      csumPart);
  reduce_sum<<<(NB * S_LEN + 255) / 256, 256, 0, stream>>>(csumPart, colrsum);

  scale_v_t<<<dim3(64, 32, NB), dim3(32, 8), 0, stream>>>(Vb, colrsum, VsT);

  out_gemm<<<dim3(16, 8, NB), 256, 0, stream>>>(Pb, VsT, out);
}
